// AxisAttention_55929064129351
// MI455X (gfx1250) — matrix operand order repaired and checked by reading, not compiled, not run on hardware
//
#include <hip/hip_runtime.h>

// ---------------------------------------------------------------------------
// AxisAttention for MI455X (gfx1250, wave32, WMMA).
//   x: (8, 256, 128, 128) fp32 -> attention along h axis.
//   B=8, C=256, H=W=128, HEADS=4, head_dim=64, seq S=H=128, batch BW=b*w=1024.
// Strategy: bf16 WMMA (V_WMMA_F32_16X16X32_BF16), f32 accumulation.
//  - A-operands staged in LDS once per block (no re-read amplification)
//  - WMMA fragments = two 16-byte vector loads per lane (K-contiguous runs)
//  - Register double-buffering of fragments across k-steps
//  - Bulk LDS staging uses GLOBAL_LOAD_ASYNC_TO_LDS_B128 (ASYNCcnt path)
// ---------------------------------------------------------------------------

typedef __attribute__((ext_vector_type(16))) __bf16          v16bf;
typedef __attribute__((ext_vector_type(16))) unsigned short  u16x16;
typedef __attribute__((ext_vector_type(8)))  unsigned short  u16x8;
typedef __attribute__((ext_vector_type(8)))  float           v8f;
typedef __attribute__((ext_vector_type(4)))  float           f32x4;

// Sizes
#define NC     256        // channels
#define NH     128        // h (sequence axis)
#define NW     128        // w
#define NHEAD  4
#define HD     64         // head dim
#define BW     1024       // b*w
#define N1     768        // 3*NC

// Workspace layout (ushort/bf16 elements). Q/K: [bh, s, d]; V: [bh, d, s].
#define OFF_Q     0ull
#define OFF_K     33554432ull
#define OFF_V     67108864ull
#define OFF_O     100663296ull   // O: [m, 256]
#define OFF_WQT   134217728ull   // Wqkv^T bf16: [768, 256]
#define OFF_WPT   134414336ull   // Wproj^T bf16: [256, 256]

// fp32 -> bf16 (round to nearest even), raw u16 bits
__device__ __forceinline__ unsigned short f2bf(float f) {
    unsigned u = __float_as_uint(f);
    u += 0x7FFFu + ((u >> 16) & 1u);
    return (unsigned short)(u >> 16);
}

// A fragment (16-bit 16x32): lane elements = 16 consecutive K at k0
__device__ __forceinline__ v16bf load_fragA(const unsigned short* row, int k0) {
    u16x8 lo = *(const u16x8*)(row + k0);
    u16x8 hi = *(const u16x8*)(row + k0 + 8);
    u16x16 v = __builtin_shufflevector(lo, hi, 0,1,2,3,4,5,6,7,8,9,10,11,12,13,14,15);
    return __builtin_bit_cast(v16bf, v);
}
// B fragment (16-bit 32x16, N-major rows): 16 consecutive K at k0
__device__ __forceinline__ v16bf load_fragB(const unsigned short* row, int k0) {
    u16x8 lo = *(const u16x8*)(row + k0);
    u16x8 hi = *(const u16x8*)(row + k0 + 8);
    u16x16 v = __builtin_shufflevector(lo, hi, 0,1,2,3,4,5,6,7,8,9,10,11,12,13,14,15);
    return __builtin_bit_cast(v16bf, v);
}

__device__ __forceinline__ v8f wmma_bf16(v16bf a, v16bf b, v8f c) {
    return __builtin_amdgcn_wmma_f32_16x16x32_bf16(false, a, false, b,
                                                   (short)0, c, false, false);
}

// Async bulk copy global -> LDS (16 bytes per lane), tracked by ASYNCcnt.
// LDS dest VGPR = AS3 byte offset = low 32 bits of the generic pointer.
__device__ __forceinline__ void async_copy_b128(void* lds_dst, const void* gsrc) {
    asm volatile("global_load_async_to_lds_b128 %0, %1, off"
                 :: "v"((unsigned)(unsigned long long)lds_dst),
                    "v"((unsigned long long)gsrc)
                 : "memory");
}
__device__ __forceinline__ void wait_async0() {
    asm volatile("s_wait_asynccnt 0x0" ::: "memory");
}

// ---------------------------------------------------------------------------
// Kernel 0: transpose + convert weights to bf16 (tiny; L2-resident afterwards).
// ---------------------------------------------------------------------------
__global__ void __launch_bounds__(256)
wprep_kernel(const float* __restrict__ Wqkv, const float* __restrict__ Wproj,
             unsigned short* __restrict__ ws) {
    const int i = blockIdx.x * 256 + threadIdx.x;   // 0 .. 262143
    if (i < N1 * NC) {
        const int n = i >> 8, k = i & 255;
        ws[OFF_WQT + i] = f2bf(Wqkv[(long)k * N1 + n]);
    } else {
        const int j = i - N1 * NC;
        const int n = j >> 8, k = j & 255;
        ws[OFF_WPT + j] = f2bf(Wproj[(long)k * NC + n]);
    }
}

// ---------------------------------------------------------------------------
// Kernel 1: QKV = x_r @ Wqkv + bqkv.
// One block per M-tile (16 rows = 16 consecutive w for fixed (b,h)).
// Stage the 16x256 x-slab once in LDS (coalesced f32x4 loads, bf16 convert),
// then 8 waves x 6 N-tiles reuse it. Fragments double-buffered across k.
// ---------------------------------------------------------------------------
__global__ void __launch_bounds__(256)
qkv_gemm_kernel(const float* __restrict__ x, const float* __restrict__ bqkv,
                unsigned short* __restrict__ ws) {
    __shared__ unsigned short sA[16 * NC];   // 8 KB: [row(w)][k] K-contiguous

    const int tid  = threadIdx.x;
    const int wave = tid >> 5;
    const int lane = tid & 31;
    const int lm   = lane & 15;
    const int hi   = lane >> 4;

    const int mt = blockIdx.x;            // 0..8191: (b, h, wt)
    const int b  = mt >> 10;
    const int h  = (mt >> 3) & 127;
    const int wt = mt & 7;

    // ---- stage A: thread t handles channel k = t, 16 contiguous w -------
    {
        const int k = tid;
        const float* xg = x + (long)b * (NC * NH * NW) + (long)k * (NH * NW)
                            + h * NW + wt * 16;
#pragma unroll
        for (int j = 0; j < 4; ++j) {
            const f32x4 v = *(const f32x4*)(xg + j * 4);
#pragma unroll
            for (int e = 0; e < 4; ++e)
                sA[(j * 4 + e) * NC + k] = f2bf(v[e]);
        }
    }
    __syncthreads();

    // ---- 8 waves x 6 N-tiles; double-buffered fragments ------------------
    const int a0 = hi << 4, b0 = hi << 4;
    const unsigned short* arow = sA + lm * NC;
    const unsigned short* brow[6];
#pragma unroll
    for (int j = 0; j < 6; ++j)
        brow[j] = ws + OFF_WQT + (long)((wave * 6 + j) * 16 + lm) * NC;

    v8f acc[6] = {};
    v16bf aF = load_fragA(arow, a0);
    v16bf bF[6];
#pragma unroll
    for (int j = 0; j < 6; ++j) bF[j] = load_fragB(brow[j], b0);

#pragma unroll
    for (int kk = 0; kk < 8; ++kk) {      // K = 256 in chunks of 32
        v16bf aN, bN[6];
        if (kk < 7) {                     // prefetch next k-step's fragments
            aN = load_fragA(arow, (kk + 1) * 32 + a0);
#pragma unroll
            for (int j = 0; j < 6; ++j)
                bN[j] = load_fragB(brow[j], (kk + 1) * 32 + b0);
        }
#pragma unroll
        for (int j = 0; j < 6; ++j)
            acc[j] = wmma_bf16(aF, bF[j], acc[j]);
        aF = aN;
#pragma unroll
        for (int j = 0; j < 6; ++j) bF[j] = bN[j];
    }

    // ---- epilogue: bias + bf16, scatter into Q/K [bh,s,d], V [bh,d,s] ----
#pragma unroll
    for (int j = 0; j < 6; ++j) {
        const int nt    = wave * 6 + j;
        const int ng    = nt * 16 + lm;
        const int which = ng >> 8;            // uniform per tile
        const int head  = (ng >> 6) & 3;      // uniform per tile
        const int d     = ng & 63;
        const float bias = bqkv[ng];
#pragma unroll
        for (int r = 0; r < 8; ++r) {
            const int wrow = wt * 16 + r + (hi << 3);
            const int bh   = (b * NW + wrow) * NHEAD + head;
            const float v  = acc[j][r] + bias;
            unsigned long long off;
            if (which == 2)   // V transposed: [bh, d, s]
                off = OFF_V + ((unsigned long long)bh * HD + d) * NH + h;
            else              // Q/K: [bh, s, d]
                off = (which ? OFF_K : OFF_Q) +
                      ((unsigned long long)bh * NH + h) * HD + d;
            ws[off] = f2bf(v);
        }
    }
}

// ---------------------------------------------------------------------------
// Kernel 2: per-(bw,head) attention, fully LDS-resident.
// Async-stage Q,K,Vt (3x16KB) -> scores WMMA -> LDS f32 (64KB) -> softmax ->
// bf16 probs (32KB) -> PV WMMA. 144KB dynamic LDS, 8 waves.
// ---------------------------------------------------------------------------
__global__ void __launch_bounds__(256)
attn_kernel(const unsigned short* __restrict__ ws, unsigned short* __restrict__ obuf) {
    extern __shared__ char smem[];
    float*          sS = (float*)smem;                        // 64 KB
    unsigned short* sP = (unsigned short*)(smem + 65536);     // 32 KB
    unsigned short* sQ = (unsigned short*)(smem + 98304);     // 16 KB [s][d]
    unsigned short* sK = (unsigned short*)(smem + 114688);    // 16 KB [s][d]
    unsigned short* sV = (unsigned short*)(smem + 131072);    // 16 KB [d][s]

    const int bh   = blockIdx.x;          // bw*4 + head
    const int tid  = threadIdx.x;
    const int wave = tid >> 5;            // m-tile 0..7
    const int lane = tid & 31;
    const int lm   = lane & 15;
    const int hi   = lane >> 4;

    const unsigned short* Qg = ws + OFF_Q + (unsigned long long)bh * (NH * HD);
    const unsigned short* Kg = ws + OFF_K + (unsigned long long)bh * (NH * HD);
    const unsigned short* Vg = ws + OFF_V + (unsigned long long)bh * (HD * NH);

    // ---- async stage Q/K/Vt: 3 x 8192 ushorts, 32 per thread -------------
#pragma unroll
    for (int j = 0; j < 4; ++j) {
        const int o = tid * 32 + j * 8;
        async_copy_b128(sQ + o, Qg + o);
        async_copy_b128(sK + o, Kg + o);
        async_copy_b128(sV + o, Vg + o);
    }
    wait_async0();
    __syncthreads();

    // ---- scores S = (Q K^T) * scale ------------------------------------
    const int mt = wave;
    const unsigned short* qrow = sQ + (mt * 16 + lm) * HD;
    for (int ct = 0; ct < 8; ++ct) {
        const unsigned short* krow = sK + (ct * 16 + lm) * HD;   // B[d,n]=K[n,d]
        v8f acc = {};
#pragma unroll
        for (int kk = 0; kk < 2; ++kk) {  // d = 64 in chunks of 32
            const v16bf a  = load_fragA(qrow, kk * 32 + (hi << 4));
            const v16bf bb = load_fragB(krow, kk * 32 + (hi << 4));
            acc = wmma_bf16(a, bb, acc);
        }
#pragma unroll
        for (int r = 0; r < 8; ++r)
            sS[(mt * 16 + r + (hi << 3)) * NH + ct * 16 + lm] = acc[r] * 0.125f;
    }
    __syncthreads();

    // ---- softmax over each row of 128; emit bf16 probs ------------------
    if (tid < NH) {
        float* row = sS + tid * NH;
        float mx = row[0];
        for (int i = 1; i < NH; ++i) mx = fmaxf(mx, row[i]);
        float sum = 0.0f;
        for (int i = 0; i < NH; ++i) { float ev = __expf(row[i] - mx); row[i] = ev; sum += ev; }
        const float inv = 1.0f / sum;
        for (int i = 0; i < NH; ++i) sP[tid * NH + i] = f2bf(row[i] * inv);
    }
    __syncthreads();

    // ---- O = P @ V  (both operands in LDS) ------------------------------
    const int bw   = bh >> 2;
    const int head = bh & 3;
    const unsigned short* prow = sP + (mt * 16 + lm) * NH;
    for (int nt = 0; nt < 4; ++nt) {      // d = 64 -> 4 n-tiles
        const unsigned short* vrow = sV + (nt * 16 + lm) * NH;   // B[s,d]=Vt[d,s]
        v8f acc = {};
#pragma unroll
        for (int kk = 0; kk < 4; ++kk) {  // s = 128 in chunks of 32
            const v16bf a  = load_fragA(prow, kk * 32 + (hi << 4));
            const v16bf bb = load_fragB(vrow, kk * 32 + (hi << 4));
            acc = wmma_bf16(a, bb, acc);
        }
#pragma unroll
        for (int r = 0; r < 8; ++r) {
            const int s = mt * 16 + r + (hi << 3);
            const int d = nt * 16 + lm;
            obuf[((unsigned long long)(bw * NH + s)) * NC + head * HD + d] = f2bf(acc[r]);
        }
    }
}

// ---------------------------------------------------------------------------
// Kernel 3: out = O @ Wproj + bproj, scatter back to (b, c, h, w) fp32.
// One block per M-tile; async-stage the 16x256 O-slab (8KB LDS);
// 8 waves x 2 N-tiles, double-buffered fragments.
// ---------------------------------------------------------------------------
__global__ void __launch_bounds__(256)
proj_gemm_kernel(const unsigned short* __restrict__ obuf,
                 const unsigned short* __restrict__ wpt,
                 const float* __restrict__ bproj, float* __restrict__ out) {
    __shared__ unsigned short sA[16 * NC];   // 8 KB

    const int tid  = threadIdx.x;
    const int wave = tid >> 5;
    const int lane = tid & 31;
    const int lm   = lane & 15;
    const int hi   = lane >> 4;

    const int mt = blockIdx.x;            // 0..8191

    // ---- async stage A: straight 8KB copy (16 ushorts per thread) -------
    {
        const int o = tid * 16;
        const unsigned short* src = obuf + (long)mt * (16 * NC) + o;
        async_copy_b128(sA + o,     src);
        async_copy_b128(sA + o + 8, src + 8);
    }
    wait_async0();
    __syncthreads();

    const int a0 = hi << 4, b0 = hi << 4;
    const unsigned short* arow = sA + lm * NC;
    const unsigned short* brow[2];
#pragma unroll
    for (int j = 0; j < 2; ++j)
        brow[j] = wpt + (long)((wave * 2 + j) * 16 + lm) * NC;

    v8f acc[2] = {};
    v16bf aF = load_fragA(arow, a0);
    v16bf bF[2];
#pragma unroll
    for (int j = 0; j < 2; ++j) bF[j] = load_fragB(brow[j], b0);

#pragma unroll
    for (int kk = 0; kk < 8; ++kk) {      // K = 256 in chunks of 32
        v16bf aN, bN[2];
        if (kk < 7) {
            aN = load_fragA(arow, (kk + 1) * 32 + a0);
#pragma unroll
            for (int j = 0; j < 2; ++j)
                bN[j] = load_fragB(brow[j], (kk + 1) * 32 + b0);
        }
#pragma unroll
        for (int j = 0; j < 2; ++j)
            acc[j] = wmma_bf16(aF, bF[j], acc[j]);
        aF = aN;
#pragma unroll
        for (int j = 0; j < 2; ++j) bF[j] = bN[j];
    }

#pragma unroll
    for (int j = 0; j < 2; ++j) {
        const int c    = (wave * 2 + j) * 16 + lm;
        const float bias = bproj[c];
#pragma unroll
        for (int r = 0; r < 8; ++r) {
            const int Mrow = mt * 16 + r + (hi << 3);  // m = b*16384 + w*128 + h
            const int h    = Mrow & 127;
            const int w    = (Mrow >> 7) & 127;
            const int b    = Mrow >> 14;
            out[(long)b * (NC * NH * NW) + (long)c * (NH * NW) + h * NW + w]
                = acc[j][r] + bias;
        }
    }
}

// ---------------------------------------------------------------------------
extern "C" void kernel_launch(void* const* d_in, const int* in_sizes, int n_in,
                              void* d_out, int out_size, void* d_ws, size_t ws_size,
                              hipStream_t stream) {
    const float* x     = (const float*)d_in[0];
    const float* Wqkv  = (const float*)d_in[1];
    const float* bqkv  = (const float*)d_in[2];
    const float* Wproj = (const float*)d_in[3];
    const float* bproj = (const float*)d_in[4];
    float* out = (float*)d_out;
    unsigned short* ws = (unsigned short*)d_ws;   // ~257 MiB used

    // K0: weight transpose + bf16 convert
    wprep_kernel<<<1024, 256, 0, stream>>>(Wqkv, Wproj, ws);
    // K1: one block per M-tile (8192), covers all 48 N-tiles
    qkv_gemm_kernel<<<8192, 256, 0, stream>>>(x, bqkv, ws);
    // K2: one block per (bw, head); 144 KB dynamic LDS
    attn_kernel<<<BW * NHEAD, 256, 147456, stream>>>(ws, ws + OFF_O);
    // K3: one block per M-tile (8192), covers all 16 N-tiles
    proj_gemm_kernel<<<8192, 256, 0, stream>>>(ws + OFF_O, ws + OFF_WPT, bproj, out);
}
